// ScaledDotProductAttentionMemory_31877247271230
// MI455X (gfx1250) — matrix operand order repaired and checked by reading, not compiled, not run on hardware
//
#include <hip/hip_runtime.h>
#include <hip/hip_bf16.h>

typedef unsigned short u16;
typedef __attribute__((ext_vector_type(16))) __bf16 v16b;
typedef __attribute__((ext_vector_type(8)))  float  v8f;
typedef unsigned int u32x4 __attribute__((ext_vector_type(4)));
typedef float f32x4 __attribute__((ext_vector_type(4)));
typedef u16 u16x4 __attribute__((ext_vector_type(4)));

#define D_MODEL 1024
#define DK      64
#define H_      16
#define B_      8
#define NQ_     1024
#define NK_     1024
#define M_      49
#define NKM     1073
#define NKM_PAD 1088   // 17 * 64

__device__ __forceinline__ u16 bf16_rne(float f) {
  unsigned int u = __float_as_uint(f);
  u += 0x7FFFu + ((u >> 16) & 1u);
  return (u16)(u >> 16);
}

union FragU {
  u32x4 q[2];
  u16   u[16];
  v16b  v;
};

// 16 bf16 elements from two 16-byte chunks
__device__ __forceinline__ v16b frag_u16(const u16* p0, const u16* p1) {
  FragU f;
  f.q[0] = *(const u32x4*)p0;
  f.q[1] = *(const u32x4*)p1;
  return f.v;
}

// 16 bf16 elements converted on the fly from two 8-float chunks (fallback path)
__device__ __forceinline__ v16b frag_f32(const float* p0, const float* p1) {
  FragU f;
#pragma unroll
  for (int i = 0; i < 8; ++i) f.u[i] = bf16_rne(p0[i]);
#pragma unroll
  for (int i = 0; i < 8; ++i) f.u[8 + i] = bf16_rne(p1[i]);
  return f.v;
}

__device__ __forceinline__ v16b ones_frag() {
  FragU f;
#pragma unroll
  for (int i = 0; i < 16; ++i) f.u[i] = 0x3F80;  // bf16 1.0
  return f.v;
}

__device__ __forceinline__ v8f wmma_bf16(v16b a, v16b b, v8f c) {
  return __builtin_amdgcn_wmma_f32_16x16x32_bf16(false, a, false, b, (short)0, c,
                                                 false, false);
}

// all-reduce max over each 16-lane row via DPP16 row_ror butterflies (pure VALU,
// co-executes with the matrix pipe; no ds_bpermute traffic)
__device__ __forceinline__ float rowmax16(float x) {
  int y;
  y = __builtin_amdgcn_update_dpp(0, __float_as_int(x), 0x121, 0xf, 0xf, true);
  x = fmaxf(x, __int_as_float(y));
  y = __builtin_amdgcn_update_dpp(0, __float_as_int(x), 0x122, 0xf, 0xf, true);
  x = fmaxf(x, __int_as_float(y));
  y = __builtin_amdgcn_update_dpp(0, __float_as_int(x), 0x124, 0xf, 0xf, true);
  x = fmaxf(x, __int_as_float(y));
  y = __builtin_amdgcn_update_dpp(0, __float_as_int(x), 0x128, 0xf, 0xf, true);
  x = fmaxf(x, __int_as_float(y));
  return x;
}

__global__ void cvt_f32_bf16_v4(const f32x4* __restrict__ s, u16x4* __restrict__ d,
                                int n4) {
  int i = blockIdx.x * blockDim.x + threadIdx.x;
  int stride = gridDim.x * blockDim.x;
  for (; i < n4; i += stride) {
    f32x4 x = s[i];
    u16x4 y;
    y[0] = bf16_rne(x[0]); y[1] = bf16_rne(x[1]);
    y[2] = bf16_rne(x[2]); y[3] = bf16_rne(x[3]);
    d[i] = y;
  }
}

// C[row, col] = sum_k A[row,k] * W[col,k] + bias[col]
// A: nrows x 1024 (f32 or bf16, row-major), W: 1024 x 1024 bf16 row-major.
// WG = 256 thr (8 waves), WG tile 64(M) x 256(N); wave tile 32 x 64.
// LAYOUT 0: f32 out, dst[row*1024+col]
// LAYOUT 1: bf16 Q   (b,h,q,dk)
// LAYOUT 2: bf16 K   (b,h,koff+r,dk), keys padded to 1088
// LAYOUT 3: bf16 V^T (b,h,dv,koff+r)
template <bool A_BF16, int LAYOUT>
__global__ __launch_bounds__(256) void gemm_wmma(const void* __restrict__ Ap,
                                                 const u16* __restrict__ W,
                                                 const float* __restrict__ bias,
                                                 void* __restrict__ dstp,
                                                 int nrows, int koff, int rpb) {
  const int lane  = threadIdx.x & 31;
  const int wv    = threadIdx.x >> 5;
  const int n16   = lane & 15;   // A-frag row / B-frag col / C-frag col
  const int kh    = lane >> 4;   // K-half selector; C rows = kh*8 + r
  const int mbase = blockIdx.x * 64 + (wv & 1) * 32;
  const int nbase = blockIdx.y * 256 + (wv >> 1) * 64;

  const float* Af = (const float*)Ap;
  const u16*   Ab = (const u16*)Ap;

  const v8f zero8 = {0, 0, 0, 0, 0, 0, 0, 0};
  v8f c[2][4];
#pragma unroll
  for (int i = 0; i < 2; ++i)
#pragma unroll
    for (int j = 0; j < 4; ++j) c[i][j] = zero8;

  int arow[2];
#pragma unroll
  for (int i = 0; i < 2; ++i) {
    int r = mbase + i * 16 + n16;
    arow[i] = r < nrows ? r : nrows - 1;  // clamp; bad rows never stored
  }

  for (int k = 0; k < D_MODEL; k += 32) {
    v16b a[2], bb[4];
#pragma unroll
    for (int i = 0; i < 2; ++i) {
      if (A_BF16) {
        const u16* p = Ab + (size_t)arow[i] * D_MODEL + k;
        a[i] = frag_u16(p + kh * 8, p + 16 + kh * 8);
      } else {
        const float* p = Af + (size_t)arow[i] * D_MODEL + k;
        a[i] = frag_f32(p + kh * 8, p + 16 + kh * 8);
      }
    }
#pragma unroll
    for (int j = 0; j < 4; ++j) {
      const u16* p = W + (size_t)(nbase + j * 16 + n16) * D_MODEL + k + kh * 8;
      bb[j] = frag_u16(p, p + 16);
    }
#pragma unroll
    for (int i = 0; i < 2; ++i)
#pragma unroll
      for (int j = 0; j < 4; ++j) c[i][j] = wmma_bf16(a[i], bb[j], c[i][j]);
  }

#pragma unroll
  for (int i = 0; i < 2; ++i) {
#pragma unroll
    for (int j = 0; j < 4; ++j) {
      const int col = nbase + j * 16 + n16;
      const float bj = bias[col];
#pragma unroll
      for (int r = 0; r < 8; ++r) {
        const int row = mbase + i * 16 + kh * 8 + r;
        if (row >= nrows) continue;
        const float v = c[i][j][r] + bj;
        if (LAYOUT == 0) {
          ((float*)dstp)[(size_t)row * D_MODEL + col] = v;
        } else {
          u16* dst = (u16*)dstp;
          const u16 hv = bf16_rne(v);
          const int h = col >> 6, cc = col & 63;
          if (LAYOUT == 1) {
            const int b = row >> 10, q = row & 1023;
            dst[(((size_t)b * H_ + h) * NQ_ + q) * DK + cc] = hv;
          } else if (LAYOUT == 2) {
            const int b = row / rpb, rr = row - b * rpb;
            dst[(((size_t)b * H_ + h) * NKM_PAD + (koff + rr)) * DK + cc] = hv;
          } else {
            const int b = row / rpb, rr = row - b * rpb;
            dst[(((size_t)b * H_ + h) * DK + cc) * NKM_PAD + (koff + rr)] = hv;
          }
        }
      }
    }
  }
}

// Flash attention: block 256 = 8 waves; each wave owns a 16-row Q tile and
// streams keys in blocks of 64 (amortizes the O-rescale VALU work).
// grid.x = B*H, grid.y = NQ/128.
__global__ __launch_bounds__(256) void flash_attn(const u16* __restrict__ Qp,
                                                  const u16* __restrict__ Kp,
                                                  const u16* __restrict__ Vt,
                                                  const unsigned char* __restrict__ mask,
                                                  u16* __restrict__ Og) {
  __shared__ u16 lds[8 * 16 * 64];  // per-wave 16x64 bf16 P tile (16 KB/WG)
  const int lane = threadIdx.x & 31;
  const int wv   = threadIdx.x >> 5;
  const int n16  = lane & 15;
  const int kh   = lane >> 4;
  const int bh   = blockIdx.x;  // b*16 + h
  const int b    = bh >> 4, h = bh & 15;
  const int qbase = blockIdx.y * 128 + wv * 16;
  u16* pw = lds + wv * (16 * 64);

  const u16* Qb = Qp + ((size_t)bh * NQ_ + qbase) * DK;
  const u16* Kb = Kp + (size_t)bh * NKM_PAD * DK;
  const u16* Vb = Vt + (size_t)bh * DK * NKM_PAD;
  const unsigned char* mk = mask + ((size_t)bh * NQ_ + qbase + kh * 8) * NK_;

  // Q A-fragments for dk 0..31 and 32..63 (reused across all key blocks)
  const u16* qrow = Qb + (size_t)n16 * DK;
  const v16b aq0 = frag_u16(qrow + kh * 8, qrow + 16 + kh * 8);
  const v16b aq1 = frag_u16(qrow + 32 + kh * 8, qrow + 48 + kh * 8);
  const v16b vones = ones_frag();

  const v8f zero8 = {0, 0, 0, 0, 0, 0, 0, 0};
  float mrow[8];
  v8f o[4];
  v8f o4 = zero8;  // online-softmax denominator l, computed by WMMA (P x ones)
#pragma unroll
  for (int r = 0; r < 8; ++r) mrow[r] = -1e30f;
#pragma unroll
  for (int t = 0; t < 4; ++t) o[t] = zero8;

  // exp2-domain softmax: fold log2(e) into the 1/sqrt(dk) scale
  const float scale2 = 0.125f * 1.44269504088896340736f;

  for (int j0 = 0; j0 < NKM_PAD; j0 += 64) {
    v8f s[4];
    const bool mblock = (j0 < NK_);  // uniform per block (64 | 1024)
#pragma unroll
    for (int qtr = 0; qtr < 4; ++qtr) {
      const int key = j0 + qtr * 16 + n16;
      const u16* krow = Kb + (size_t)key * DK;
      v16b kf0 = frag_u16(krow + kh * 8, krow + 16 + kh * 8);
      v16b kf1 = frag_u16(krow + 32 + kh * 8, krow + 48 + kh * 8);
      unsigned char mv[8];
      if (mblock) {
#pragma unroll
        for (int r = 0; r < 8; ++r) mv[r] = mk[(size_t)r * NK_ + key];  // clauses
      }
      v8f acc = zero8;
      acc = wmma_bf16(aq0, kf0, acc);
      acc = wmma_bf16(aq1, kf1, acc);
#pragma unroll
      for (int r = 0; r < 8; ++r) {
        const float v = acc[r] * scale2;
        const bool dead = mblock ? (mv[r] != 0) : (key >= NKM);
        s[qtr][r] = dead ? -1e30f : v;
      }
    }
    // online softmax, per row (rows live in VGPR slot r of each 16-lane half)
#pragma unroll
    for (int r = 0; r < 8; ++r) {
      const float bm =
          rowmax16(fmaxf(fmaxf(s[0][r], s[1][r]), fmaxf(s[2][r], s[3][r])));
      const float nm = fmaxf(mrow[r], bm);
      const float alpha = exp2f(mrow[r] - nm);
      mrow[r] = nm;
      float p[4];
#pragma unroll
      for (int qtr = 0; qtr < 4; ++qtr)
        p[qtr] = (s[qtr][r] <= -1e29f) ? 0.f : exp2f(s[qtr][r] - nm);
      o4[r] *= alpha;
#pragma unroll
      for (int t = 0; t < 4; ++t) o[t][r] *= alpha;
      // P (C-layout) -> LDS (row-major 16x64 bf16), wave-private region
#pragma unroll
      for (int qtr = 0; qtr < 4; ++qtr)
        pw[(kh * 8 + r) * 64 + qtr * 16 + n16] = bf16_rne(p[qtr]);
    }
    asm volatile("" ::: "memory");  // keep DS write -> read order (HW in-order per wave)
    // re-read P as A-fragments (16 q rows x 64 keys)
    const u16* prow = pw + n16 * 64;
    v16b pa0 = frag_u16(prow + kh * 8, prow + 16 + kh * 8);
    v16b pa1 = frag_u16(prow + 32 + kh * 8, prow + 48 + kh * 8);
#pragma unroll
    for (int t = 0; t < 4; ++t) {
      const u16* vrow = Vb + (size_t)(t * 16 + n16) * NKM_PAD + j0 + kh * 8;
      v16b vf0 = frag_u16(vrow, vrow + 16);
      v16b vf1 = frag_u16(vrow + 32, vrow + 48);
      o[t] = wmma_bf16(pa0, vf0, o[t]);
      o[t] = wmma_bf16(pa1, vf1, o[t]);
    }
    o4 = wmma_bf16(pa0, vones, o4);  // l += rowsum(P), rescaled like o[]
    o4 = wmma_bf16(pa1, vones, o4);
  }

#pragma unroll
  for (int r = 0; r < 8; ++r) {
    const float l = o4[r];
    const float inv = l > 0.f ? 1.f / l : 0.f;
    const int q = qbase + kh * 8 + r;
    u16* orow = Og + ((size_t)b * NQ_ + q) * D_MODEL + h * DK;
#pragma unroll
    for (int t = 0; t < 4; ++t) orow[t * 16 + n16] = bf16_rne(o[t][r] * inv);
  }
}

extern "C" void kernel_launch(void* const* d_in, const int* in_sizes, int n_in,
                              void* d_out, int out_size, void* d_ws, size_t ws_size,
                              hipStream_t stream) {
  const float* queries = (const float*)d_in[0];
  const float* keys    = (const float*)d_in[1];
  const float* values  = (const float*)d_in[2];
  const unsigned char* amask = (const unsigned char*)d_in[3];  // jnp bool -> 1 byte
  const float* key_grid = (const float*)d_in[4];
  // d_in[5] value_grid intentionally unused (reference uses key_grid for gv)
  const float* Wq = (const float*)d_in[6];  const float* bq  = (const float*)d_in[7];
  const float* Wk = (const float*)d_in[8];  const float* bk  = (const float*)d_in[9];
  const float* Wgk= (const float*)d_in[10]; const float* bgk = (const float*)d_in[11];
  const float* Wv = (const float*)d_in[12]; const float* bv  = (const float*)d_in[13];
  const float* Wgv= (const float*)d_in[14]; const float* bgv = (const float*)d_in[15];
  const float* Wo = (const float*)d_in[16]; const float* bo  = (const float*)d_in[17];
  (void)in_sizes; (void)n_in; (void)out_size;

  u16* ws = (u16*)d_ws;
  size_t off = 0;
  auto alloc = [&](size_t n) { u16* p = ws + off; off += n; return p; };
  const size_t WSZ  = (size_t)D_MODEL * D_MODEL;       // 1048576
  const size_t NXQ  = (size_t)B_ * NQ_ * D_MODEL;      // 8388608
  const size_t NXG  = (size_t)B_ * M_ * D_MODEL;       // 401408
  u16* Wqb  = alloc(WSZ);
  u16* Wkb  = alloc(WSZ);
  u16* Wgkb = alloc(WSZ);
  u16* Wvb  = alloc(WSZ);
  u16* Wgvb = alloc(WSZ);
  u16* Wob  = alloc(WSZ);
  u16* Qp = alloc((size_t)B_ * H_ * NQ_ * DK);
  u16* Kp = alloc((size_t)B_ * H_ * NKM_PAD * DK);
  u16* Vt = alloc((size_t)B_ * H_ * NKM_PAD * DK);
  u16* Og = alloc((size_t)B_ * NQ_ * D_MODEL);
  const size_t need_act = (off + 3 * NXQ + NXG) * sizeof(u16);
  const bool preconv = ws_size >= need_act;  // host-side, deterministic
  u16 *Xq = nullptr, *Xk = nullptr, *Xv = nullptr, *Xg = nullptr;
  if (preconv) { Xq = alloc(NXQ); Xk = alloc(NXQ); Xv = alloc(NXQ); Xg = alloc(NXG); }

  dim3 blk(256);

  cvt_f32_bf16_v4<<<512, blk, 0, stream>>>((const f32x4*)Wq,  (u16x4*)Wqb,  (int)(WSZ / 4));
  cvt_f32_bf16_v4<<<512, blk, 0, stream>>>((const f32x4*)Wk,  (u16x4*)Wkb,  (int)(WSZ / 4));
  cvt_f32_bf16_v4<<<512, blk, 0, stream>>>((const f32x4*)Wgk, (u16x4*)Wgkb, (int)(WSZ / 4));
  cvt_f32_bf16_v4<<<512, blk, 0, stream>>>((const f32x4*)Wv,  (u16x4*)Wvb,  (int)(WSZ / 4));
  cvt_f32_bf16_v4<<<512, blk, 0, stream>>>((const f32x4*)Wgv, (u16x4*)Wgvb, (int)(WSZ / 4));
  cvt_f32_bf16_v4<<<512, blk, 0, stream>>>((const f32x4*)Wo,  (u16x4*)Wob,  (int)(WSZ / 4));
  if (preconv) {
    cvt_f32_bf16_v4<<<2048, blk, 0, stream>>>((const f32x4*)queries,  (u16x4*)Xq, (int)(NXQ / 4));
    cvt_f32_bf16_v4<<<2048, blk, 0, stream>>>((const f32x4*)keys,     (u16x4*)Xk, (int)(NXQ / 4));
    cvt_f32_bf16_v4<<<2048, blk, 0, stream>>>((const f32x4*)values,   (u16x4*)Xv, (int)(NXQ / 4));
    cvt_f32_bf16_v4<<<512,  blk, 0, stream>>>((const f32x4*)key_grid, (u16x4*)Xg, (int)(NXG / 4));
  }

  dim3 gproj(128, 4);  // 8192/64 x 1024/256
  dim3 ggrid(7, 4);    // ceil(392/64) x 1024/256
  if (preconv) {
    gemm_wmma<true, 1><<<gproj, blk, 0, stream>>>(Xq, Wqb,  bq,  Qp, B_ * NQ_, 0,   NQ_);
    gemm_wmma<true, 2><<<gproj, blk, 0, stream>>>(Xk, Wkb,  bk,  Kp, B_ * NK_, 0,   NK_);
    gemm_wmma<true, 2><<<ggrid, blk, 0, stream>>>(Xg, Wgkb, bgk, Kp, B_ * M_,  NK_, M_);
    gemm_wmma<true, 3><<<gproj, blk, 0, stream>>>(Xv, Wvb,  bv,  Vt, B_ * NK_, 0,   NK_);
    gemm_wmma<true, 3><<<ggrid, blk, 0, stream>>>(Xg, Wgvb, bgv, Vt, B_ * M_,  NK_, M_);
  } else {
    gemm_wmma<false, 1><<<gproj, blk, 0, stream>>>(queries,  Wqb,  bq,  Qp, B_ * NQ_, 0,   NQ_);
    gemm_wmma<false, 2><<<gproj, blk, 0, stream>>>(keys,     Wkb,  bk,  Kp, B_ * NK_, 0,   NK_);
    gemm_wmma<false, 2><<<ggrid, blk, 0, stream>>>(key_grid, Wgkb, bgk, Kp, B_ * M_,  NK_, M_);
    gemm_wmma<false, 3><<<gproj, blk, 0, stream>>>(values,   Wvb,  bv,  Vt, B_ * NK_, 0,   NK_);
    gemm_wmma<false, 3><<<ggrid, blk, 0, stream>>>(key_grid, Wgvb, bgv, Vt, B_ * M_,  NK_, M_);
  }

  dim3 gattn(B_ * H_, NQ_ / 128);
  flash_attn<<<gattn, blk, 0, stream>>>(Qp, Kp, Vt, amask, Og);

  gemm_wmma<true, 0><<<gproj, blk, 0, stream>>>(Og, Wob, bo, d_out, B_ * NQ_, 0, NQ_);
}
